// PairwiseScores_multipred_154618822962
// MI455X (gfx1250) — matrix operand order repaired and checked by reading, not compiled, not run on hardware
//
#include <hip/hip_runtime.h>

#define DIMK 256
#define HIDN 128
#define NBAT 16
#define NQ 128
#define NK 128
#define ASTRIDE 136   // bf16 elements per activation row (128 + 8 pad -> 272B pitch, bank-conflict free)

typedef __attribute__((ext_vector_type(16))) __bf16 v16bf;
typedef __attribute__((ext_vector_type(4)))  __bf16 v4bf;
typedef __attribute__((ext_vector_type(8)))  float  v8f;
typedef __attribute__((ext_vector_type(4)))  float  v4f;
typedef __attribute__((ext_vector_type(4)))  unsigned int v4u;

union FragU { v16bf f; v4u u[2]; };

// ---- A fragment (16x32 bf16) from a row-major [16][ASTRIDE] activation strip -------------
// ISA layout: lanes 0-15 (row m=lane) hold K kb..kb+7 in regs 0-3 and K kb+16..kb+23 in 4-7;
// lanes 16-31 hold the +8 K offsets.
__device__ __forceinline__ v16bf load_a_frag(const __bf16* act, int mm, int lhalf, int ks) {
  const int kb = ks * 32 + lhalf * 8;
  FragU r;
  r.u[0] = *(const v4u*)(act + mm * ASTRIDE + kb);
  r.u[1] = *(const v4u*)(act + mm * ASTRIDE + kb + 16);
  return r.f;
}

// ---- B fragment (32x16 bf16) straight from row-major bf16 weights W[n][k] ----------------
// Lanes 0-15 (col n=lane) hold K kb..kb+7 and kb+16..kb+23; lanes 16-31 hold the +8 K offsets.
__device__ __forceinline__ v16bf load_b_frag(const __bf16* W, int nt, int nn, int lhalf, int ks) {
  const int n  = nt * 16 + nn;
  const int kb = ks * 32 + lhalf * 8;
  const __bf16* p = W + n * HIDN + kb;
  FragU r;
  r.u[0] = *(const v4u*)(p);
  r.u[1] = *(const v4u*)(p + 16);
  return r.f;
}

// ---- one dense 128->128 layer with ReLU, per-wave 16-row tile, in-place capable ----------
__device__ __forceinline__ void mlp_layer(const __bf16* W, const float* bias,
                                          const __bf16* src, __bf16* dst, int lane) {
  const int lhalf = lane >> 4, nn = lane & 15, mm = lane & 15;
  v8f acc[8];
#pragma unroll
  for (int nt = 0; nt < 8; ++nt) {
    const float bv = bias[nt * 16 + nn];
#pragma unroll
    for (int v = 0; v < 8; ++v) acc[nt][v] = bv;
  }
#pragma unroll
  for (int ks = 0; ks < 4; ++ks) {
    const v16bf a = load_a_frag(src, mm, lhalf, ks);
#pragma unroll
    for (int nt = 0; nt < 8; ++nt) {
      const v16bf bfr = load_b_frag(W, nt, nn, lhalf, ks);
      acc[nt] = __builtin_amdgcn_wmma_f32_16x16x32_bf16(
          false, a, false, bfr, (short)0, acc[nt], false, false);
    }
  }
  // C layout: lane l holds column n = l%16, rows m = v + (l<16?0:8)
#pragma unroll
  for (int nt = 0; nt < 8; ++nt)
#pragma unroll
    for (int v = 0; v < 8; ++v)
      dst[(v + lhalf * 8) * ASTRIDE + nt * 16 + nn] = (__bf16)fmaxf(acc[nt][v], 0.0f);
}

// ---- 128->3 head (weights zero-padded to 16x128), masked f32 store -----------------------
__device__ __forceinline__ void mlp_head(const __bf16* W, const float* bias3,
                                         const __bf16* src, float* __restrict__ out,
                                         int b, int q0, int k0, int wave,
                                         const unsigned char* qmask, const unsigned char* dmask,
                                         int lane) {
  const int lhalf = lane >> 4, nn = lane & 15, mm = lane & 15;
  float bv = bias3[nn < 3 ? nn : 0];
  if (nn >= 3) bv = 0.0f;
  v8f acc;
#pragma unroll
  for (int v = 0; v < 8; ++v) acc[v] = bv;
#pragma unroll
  for (int ks = 0; ks < 4; ++ks) {
    const v16bf a   = load_a_frag(src, mm, lhalf, ks);
    const v16bf bfr = load_b_frag(W, 0, nn, lhalf, ks);
    acc = __builtin_amdgcn_wmma_f32_16x16x32_bf16(
        false, a, false, bfr, (short)0, acc, false, false);
  }
  if (nn < 3) {
#pragma unroll
    for (int v = 0; v < 8; ++v) {
      const int m = v + lhalf * 8;
      const int q = q0 + wave * 2 + (m >> 3);
      const int k = k0 + (m & 7);
      const bool pm = (qmask[b * NQ + q] != 0) && (dmask[b * NK + k] != 0);
      out[(((size_t)b * NQ + q) * NK + k) * 3 + nn] = pm ? acc[v] : 0.0f;
    }
  }
}

// ===== main fused kernel: one block = (batch, 16 q, 8 k) -> 128 pair rows = 8 waves ========
__global__ __launch_bounds__(256, 1) void fused_mlp(
    const float* __restrict__ qp, const float* __restrict__ dp,
    const unsigned short* __restrict__ wpack,
    const float* __restrict__ b0, const float* __restrict__ b1, const float* __restrict__ b2,
    const float* __restrict__ p1b0, const float* __restrict__ p1b1,
    const float* __restrict__ p2b0, const float* __restrict__ p2b1,
    const unsigned char* __restrict__ qmask, const unsigned char* __restrict__ dmask,
    float* __restrict__ out1, float* __restrict__ out2)
{
  __shared__ __align__(16) __bf16 actA[8][16 * ASTRIDE];  // trunk activations (per-wave strip)
  __shared__ __align__(16) __bf16 actB[8][16 * ASTRIDE];  // branch activations

  const int b    = blockIdx.z;
  const int q0   = blockIdx.y * 16;
  const int k0   = blockIdx.x * 8;
  const int wave = threadIdx.x >> 5;
  const int lane = threadIdx.x & 31;

  const __bf16* WB    = (const __bf16*)wpack;
  const __bf16* w1    = WB;
  const __bf16* w2    = WB + 16384;
  const __bf16* p1w0  = WB + 32768;
  const __bf16* p2w0  = WB + 49152;
  const __bf16* p1w1p = WB + 65536;   // 16x128, rows 3..15 zero
  const __bf16* p2w1p = WB + 67584;

  __bf16* A  = &actA[wave][0];
  __bf16* Bb = &actB[wave][0];

  // ---- h0 = relu(qp[q] + dp[k] + b0): row m -> (q = q0 + 2*wave + m/8, k = k0 + m%8) ----
  const v4f bias0 = *(const v4f*)(b0 + lane * 4);
#pragma unroll
  for (int half = 0; half < 2; ++half) {
    const int q = q0 + wave * 2 + half;
    const v4f qv = *(const v4f*)(qp + ((size_t)(b * NQ + q)) * HIDN + lane * 4);
#pragma unroll
    for (int kk = 0; kk < 8; ++kk) {
      const v4f dv = *(const v4f*)(dp + ((size_t)(b * NK + k0 + kk)) * HIDN + lane * 4);
      const v4f h = qv + dv + bias0;
      v4bf hb;
#pragma unroll
      for (int i = 0; i < 4; ++i) hb[i] = (__bf16)fmaxf(h[i], 0.0f);
      *(v4bf*)(A + (half * 8 + kk) * ASTRIDE + lane * 4) = hb;
    }
  }

  // trunk (in place: ds loads feed WMMA accs before write-back, so no extra buffer needed)
  mlp_layer(w1, b1, A, A, lane);
  mlp_layer(w2, b2, A, A, lane);

  // branch 1
  mlp_layer(p1w0, p1b0, A, Bb, lane);
  mlp_head(p1w1p, p1b1, Bb, out1, b, q0, k0, wave, qmask, dmask, lane);

  // branch 2 (trunk still intact in A)
  mlp_layer(p2w0, p2b0, A, Bb, lane);
  mlp_head(p2w1p, p2b1, Bb, out2, b, q0, k0, wave, qmask, dmask, lane);
}

// ===== prep: qp/dp projections via WMMA (f32 -> bf16 fragments in registers) ===============
__global__ __launch_bounds__(256, 1) void prep_gemm(
    const float* __restrict__ query, const float* __restrict__ doc,
    const float* __restrict__ w0, float* __restrict__ qp, float* __restrict__ dp)
{
  const int gw   = blockIdx.x * 8 + (threadIdx.x >> 5);  // 0..255 global wave id
  const int lane = threadIdx.x & 31;
  const int lhalf = lane >> 4, nn = lane & 15, mm = lane & 15;

  const bool isDoc = gw >= 128;
  const float* X = isDoc ? doc : query;                  // [2048][256]
  const float* W = w0 + (isDoc ? DIMK : 0);              // row stride 2*DIMK
  float* O       = isDoc ? dp : qp;
  const int rowBase = (gw & 127) * 16;

  v8f acc[8];
#pragma unroll
  for (int nt = 0; nt < 8; ++nt)
#pragma unroll
    for (int v = 0; v < 8; ++v) acc[nt][v] = 0.0f;

#pragma unroll 2
  for (int ks = 0; ks < 8; ++ks) {
    // A fragment: convert f32 row slices to bf16 in registers
    const int kb = ks * 32 + lhalf * 8;
    const float* xr = X + (size_t)(rowBase + mm) * DIMK + kb;
    const v4f f0 = *(const v4f*)(xr);
    const v4f f1 = *(const v4f*)(xr + 4);
    const v4f f2 = *(const v4f*)(xr + 16);
    const v4f f3 = *(const v4f*)(xr + 20);
    v16bf a;
#pragma unroll
    for (int i = 0; i < 4; ++i) {
      a[i]      = (__bf16)f0[i];
      a[4 + i]  = (__bf16)f1[i];
      a[8 + i]  = (__bf16)f2[i];
      a[12 + i] = (__bf16)f3[i];
    }
#pragma unroll
    for (int nt = 0; nt < 8; ++nt) {
      const int n   = nt * 16 + nn;
      const int kbb = ks * 32 + lhalf * 8;
      const float* wr = W + (size_t)n * (2 * DIMK) + kbb;
      const v4f g0 = *(const v4f*)(wr);
      const v4f g1 = *(const v4f*)(wr + 4);
      const v4f g2 = *(const v4f*)(wr + 16);
      const v4f g3 = *(const v4f*)(wr + 20);
      v16bf bfr;
#pragma unroll
      for (int i = 0; i < 4; ++i) {
        bfr[i]      = (__bf16)g0[i];
        bfr[4 + i]  = (__bf16)g1[i];
        bfr[8 + i]  = (__bf16)g2[i];
        bfr[12 + i] = (__bf16)g3[i];
      }
      acc[nt] = __builtin_amdgcn_wmma_f32_16x16x32_bf16(
          false, a, false, bfr, (short)0, acc[nt], false, false);
    }
  }
#pragma unroll
  for (int nt = 0; nt < 8; ++nt)
#pragma unroll
    for (int v = 0; v < 8; ++v)
      O[(size_t)(rowBase + v + lhalf * 8) * HIDN + nt * 16 + nn] = acc[nt][v];
}

// ===== prep: convert the four 128x128 weights + zero-padded 16x128 heads to bf16 ===========
__global__ void prep_cvt(const float* __restrict__ w1, const float* __restrict__ w2,
                         const float* __restrict__ p1w0, const float* __restrict__ p2w0,
                         const float* __restrict__ p1w1, const float* __restrict__ p2w1,
                         unsigned short* __restrict__ o)
{
  const int t = blockIdx.x * 256 + threadIdx.x;
  __bf16* ob = (__bf16*)o;
  if (t < 65536) {
    const float* srcs[4] = {w1, w2, p1w0, p2w0};
    ob[t] = (__bf16)srcs[t >> 14][t & 16383];
  } else if (t < 69632) {
    const int j = t - 65536;
    const int which = j >> 11, jj = j & 2047;
    const int row = jj >> 7, col = jj & 127;
    const float* s = which ? p2w1 : p1w1;
    ob[t] = (__bf16)(row < 3 ? s[row * 128 + col] : 0.0f);
  }
}

extern "C" void kernel_launch(void* const* d_in, const int* in_sizes, int n_in,
                              void* d_out, int out_size, void* d_ws, size_t ws_size,
                              hipStream_t stream) {
  const float* query = (const float*)d_in[0];
  const float* doc   = (const float*)d_in[1];
  const unsigned char* qmask = (const unsigned char*)d_in[2];
  const unsigned char* dmask = (const unsigned char*)d_in[3];
  const float* w0   = (const float*)d_in[4];
  const float* b0   = (const float*)d_in[5];
  const float* w1   = (const float*)d_in[6];
  const float* b1   = (const float*)d_in[7];
  const float* w2   = (const float*)d_in[8];
  const float* b2   = (const float*)d_in[9];
  const float* p1w0 = (const float*)d_in[10];
  const float* p1b0 = (const float*)d_in[11];
  const float* p1w1 = (const float*)d_in[12];
  const float* p1b1 = (const float*)d_in[13];
  const float* p2w0 = (const float*)d_in[14];
  const float* p2b0 = (const float*)d_in[15];
  const float* p2w1 = (const float*)d_in[16];
  const float* p2b1 = (const float*)d_in[17];

  float* qp = (float*)d_ws;               // [2048][128] f32
  float* dp = qp + 2048 * 128;            // [2048][128] f32
  unsigned short* wpack = (unsigned short*)(dp + 2048 * 128);  // bf16 weights, 69632 elems

  prep_cvt<<<272, 256, 0, stream>>>(w1, w2, p1w0, p2w0, p1w1, p2w1, wpack);
  prep_gemm<<<32, 256, 0, stream>>>(query, doc, w0, qp, dp);

  float* out1 = (float*)d_out;
  float* out2 = out1 + (size_t)NBAT * NQ * NK * 3;
  fused_mlp<<<dim3(16, 8, 16), 256, 0, stream>>>(
      qp, dp, wpack, b0, b1, b2, p1b0, p1b1, p2b0, p2b1, qmask, dmask, out1, out2);
}
